// Retention_3281355014305
// MI455X (gfx1250) — matrix operand order repaired and checked by reading, not compiled, not run on hardware
//
#include <hip/hip_runtime.h>

#define BATCH 8
#define SEQ   2048
#define DIN   512
#define DIM   512
#define KSTEP 32
#define A_ST  40   // LDS row stride (halfs) for 32-half rows: 80B, 16B-aligned, bank-spread
#define B_ST  40

typedef __attribute__((ext_vector_type(16))) _Float16 v16h;
typedef __attribute__((ext_vector_type(8)))  _Float16 v8h;
typedef __attribute__((ext_vector_type(8)))  float    v8f;
typedef __attribute__((ext_vector_type(4)))  float    v4f;
typedef __attribute__((ext_vector_type(4)))  int      v4i;

// Address-space-qualified v4i for the async-copy builtin
// (signature leaked by the round-2 diagnostic: (v4i AS1*, v4i AS3*, int, int)).
typedef __attribute__((address_space(1))) v4i v4i_g;
typedef __attribute__((address_space(3))) v4i v4i_l;

#define HAS_ASYNC_LDS __has_builtin(__builtin_amdgcn_global_load_async_to_lds_b128)

// ---------------------------------------------------------------------------
// 16-byte global -> LDS copy. Prefer the CDNA5 async path (ASYNCcnt-tracked),
// fall back to a plain vector copy if the builtin is unavailable.
// ---------------------------------------------------------------------------
__device__ __forceinline__ void cp16(_Float16* dst_lds, const _Float16* src_glob) {
#if HAS_ASYNC_LDS
  __builtin_amdgcn_global_load_async_to_lds_b128(
      (v4i_g*)src_glob, (v4i_l*)dst_lds, 0, 0);
#else
  *(v8h*)dst_lds = *(const v8h*)src_glob;
#endif
}

// Wait for this wave's async copies, then make LDS visible block-wide.
__device__ __forceinline__ void stage_join() {
#if HAS_ASYNC_LDS
#if __has_builtin(__builtin_amdgcn_s_wait_asynccnt)
  __builtin_amdgcn_s_wait_asynccnt(0);
#else
  asm volatile("s_wait_asynccnt 0x0" ::: "memory");
#endif
#endif
  __syncthreads();
}

// ---------------------------------------------------------------------------
// WMMA helper (gfx1250, wave32): D(16x16,f32) = A(16x32,f16) x B(32x16,f16)+C
// ---------------------------------------------------------------------------
__device__ __forceinline__ v8f wmma_f16(v16h a, v16h b, v8f c) {
  return __builtin_amdgcn_wmma_f32_16x16x32_f16(false, a, false, b, (short)0, c,
                                                false, false);
}

__device__ __forceinline__ v8f vzero8() {
  v8f z;
#pragma unroll
  for (int i = 0; i < 8; ++i) z[i] = 0.0f;
  return z;
}

// ---------------------------------------------------------------------------
// Cooperative copy of one K-step stage into LDS.
// A tile: 128 rows x 32 halfs, row-major (stride lda).  512 chunks, 2/thread.
// B tile:  64 rows x 32 halfs, col-major rows (stride ldb). 256 chunks, 1/thread.
// ---------------------------------------------------------------------------
__device__ __forceinline__ void stage_copy(_Float16* As, _Float16* Bs,
                                           const _Float16* Ag, int lda,
                                           const _Float16* Bg, int ldb,
                                           int tid) {
#pragma unroll
  for (int i = 0; i < 2; ++i) {
    const int c   = tid * 2 + i;
    const int row = c >> 2;
    const int ko  = (c & 3) << 3;
    cp16(As + row * A_ST + ko, Ag + (size_t)row * lda + ko);
  }
  {
    const int row = tid >> 2;
    const int ko  = (tid & 3) << 3;
    cp16(Bs + row * B_ST + ko, Bg + (size_t)row * ldb + ko);
  }
}

// A fragment from LDS: lane l -> row wave*16 + (l&15), K = {kb..kb+15}, kb = (l>>4)*16
__device__ __forceinline__ v16h lds_a_frag(const _Float16* As, int wave, int lane) {
  const int m  = wave * 16 + (lane & 15);
  const int kb = (lane >> 4) << 4;
  const _Float16* p = As + m * A_ST + kb;
  v8h lo = *(const v8h*)(p);
  v8h hi = *(const v8h*)(p + 8);
  v16h f;
#pragma unroll
  for (int i = 0; i < 8; ++i) { f[i] = lo[i]; f[i + 8] = hi[i]; }
  return f;
}

// B fragment from LDS (col-major rows): lane l -> col l&15, K from (l>>4)*16
__device__ __forceinline__ v16h lds_b_frag(const _Float16* Bs, int j, int lane) {
  const int n  = j * 16 + (lane & 15);
  const int kb = (lane >> 4) << 4;
  const _Float16* p = Bs + n * B_ST + kb;
  v8h lo = *(const v8h*)(p);
  v8h hi = *(const v8h*)(p + 8);
  v16h f;
#pragma unroll
  for (int i = 0; i < 8; ++i) { f[i] = lo[i]; f[i + 8] = hi[i]; }
  return f;
}

// A fragment straight from f32 global (convert on the fly) — projection only.
__device__ __forceinline__ v16h load_a_f32cvt(const float* __restrict__ tile,
                                              int ld, int lane) {
  const int m  = lane & 15;
  const int kb = (lane >> 4) << 4;
  const float* p = tile + (size_t)m * ld + kb;
  v4f a0 = *(const v4f*)(p);
  v4f a1 = *(const v4f*)(p + 4);
  v4f b0 = *(const v4f*)(p + 8);
  v4f b1 = *(const v4f*)(p + 12);
  v16h f;
#pragma unroll
  for (int i = 0; i < 4; ++i) {
    f[i]      = (_Float16)a0[i];
    f[i + 4]  = (_Float16)a1[i];
    f[i + 8]  = (_Float16)b0[i];
    f[i + 12] = (_Float16)b1[i];
  }
  return f;
}

// B fragment straight from col-major global — projection only.
__device__ __forceinline__ v16h load_b_col(const _Float16* __restrict__ tile,
                                           int ld, int lane) {
  const int n  = lane & 15;
  const int kb = (lane >> 4) << 4;
  return *(const v16h*)(tile + (size_t)n * ld + kb);
}

// ---------------------------------------------------------------------------
// Wt[n*DIN + k] = (f16) W[k*DIM + n]
// ---------------------------------------------------------------------------
__global__ __launch_bounds__(256) void transpose_w_kernel(
    const float* __restrict__ W, _Float16* __restrict__ Wt) {
  const int idx = blockIdx.x * 256 + threadIdx.x;
  const int n = idx >> 9;
  const int k = idx & 511;
  Wt[idx] = (_Float16)W[(size_t)k * DIM + n];
}

// ---------------------------------------------------------------------------
// cs[j] = 1/sqrt( sum_k D[j,k] )
// ---------------------------------------------------------------------------
__global__ __launch_bounds__(256) void colscale_kernel(
    const float* __restrict__ D, float* __restrict__ cs) {
  __shared__ float red[256];
  const int j = blockIdx.x;
  float s = 0.0f;
  for (int t = threadIdx.x; t < SEQ; t += 256) s += D[(size_t)j * SEQ + t];
  red[threadIdx.x] = s;
  __syncthreads();
#pragma unroll
  for (int off = 128; off > 0; off >>= 1) {
    if (threadIdx.x < off) red[threadIdx.x] += red[threadIdx.x + off];
    __syncthreads();
  }
  if (threadIdx.x == 0) cs[j] = rsqrtf(red[0]);
}

// ---------------------------------------------------------------------------
// Shared projection: P = x @ W (z selects x_q/x_k/x_v).
// Q,K stored row-major f16; V stored transposed Vt[b][d][s].
// ---------------------------------------------------------------------------
__global__ __launch_bounds__(256) void proj_kernel(
    const float* __restrict__ xq, const float* __restrict__ xk,
    const float* __restrict__ xv, const _Float16* __restrict__ Wt,
    _Float16* __restrict__ Qh, _Float16* __restrict__ Kh,
    _Float16* __restrict__ Vt) {
  const int lane = threadIdx.x & 31;
  const int wave = threadIdx.x >> 5;
  const int m0 = blockIdx.x * 128 + wave * 16;
  const int n0 = blockIdx.y * 64;
  const int z  = blockIdx.z;
  const float* X = (z == 0) ? xq : (z == 1) ? xk : xv;

  v8f acc[4];
#pragma unroll
  for (int j = 0; j < 4; ++j) acc[j] = vzero8();

  for (int kt = 0; kt < DIN; kt += KSTEP) {
    v16h a = load_a_f32cvt(X + (size_t)m0 * DIN + kt, DIN, lane);
#pragma unroll
    for (int j = 0; j < 4; ++j) {
      v16h b = load_b_col(Wt + (size_t)(n0 + j * 16) * DIN + kt, DIN, lane);
      acc[j] = wmma_f16(a, b, acc[j]);
    }
  }

  const int mo = (lane >> 4) * 8;
  const int nn = lane & 15;
  if (z != 2) {
    _Float16* O = (z == 0) ? Qh : Kh;
#pragma unroll
    for (int j = 0; j < 4; ++j)
#pragma unroll
      for (int r = 0; r < 8; ++r)
        O[(size_t)(m0 + mo + r) * DIM + (n0 + j * 16 + nn)] =
            (_Float16)acc[j][r];
  } else {
    const int b = m0 / SEQ;
    const int s = m0 % SEQ;
#pragma unroll
    for (int j = 0; j < 4; ++j)
#pragma unroll
      for (int r = 0; r < 8; ++r)
        Vt[(size_t)b * DIM * SEQ + (size_t)(n0 + j * 16 + nn) * SEQ +
           (s + mo + r)] = (_Float16)acc[j][r];
  }
}

// ---------------------------------------------------------------------------
// M[q,k] = sum_b sum_d Q[b,q,d]*K[b,k,d]; Ah = (f16) max(|M*scale*D*cs|, 1)
// LDS double-buffered pipeline: async copy stage s+1 while computing stage s.
// ---------------------------------------------------------------------------
__global__ __launch_bounds__(256) void score_kernel(
    const _Float16* __restrict__ Qh, const _Float16* __restrict__ Kh,
    const float* __restrict__ D, const float* __restrict__ cs,
    _Float16* __restrict__ Ah) {
  __shared__ _Float16 As[2][128 * A_ST];
  __shared__ _Float16 Bs[2][64 * B_ST];

  const int tid  = threadIdx.x;
  const int lane = tid & 31;
  const int wave = tid >> 5;
  const int mblk = blockIdx.x * 128;
  const int nblk = blockIdx.y * 64;

  v8f acc[4];
#pragma unroll
  for (int j = 0; j < 4; ++j) acc[j] = vzero8();

  const int NSTEP = BATCH * (DIM / KSTEP);  // 128: step s -> batch s>>4, kt (s&15)*32
  const size_t BS = (size_t)SEQ * DIM;

  stage_copy(As[0], Bs[0],
             Qh + (size_t)mblk * DIM, DIM,
             Kh + (size_t)nblk * DIM, DIM, tid);
  stage_join();

  for (int s = 0; s < NSTEP; ++s) {
    const int cur = s & 1;
    if (s + 1 < NSTEP) {
      const int sn = s + 1;
      stage_copy(As[cur ^ 1], Bs[cur ^ 1],
                 Qh + (size_t)(sn >> 4) * BS + (size_t)mblk * DIM + (sn & 15) * KSTEP, DIM,
                 Kh + (size_t)(sn >> 4) * BS + (size_t)nblk * DIM + (sn & 15) * KSTEP, DIM,
                 tid);
    }
    v16h a = lds_a_frag(As[cur], wave, lane);
#pragma unroll
    for (int j = 0; j < 4; ++j) {
      v16h bf = lds_b_frag(Bs[cur], j, lane);
      acc[j] = wmma_f16(a, bf, acc[j]);
    }
    stage_join();
  }

  const float scale = 0.04419417382415922f;  // 1/sqrt(512)
  const int m0 = mblk + wave * 16;
  const int mo = (lane >> 4) * 8;
  const int nn = lane & 15;
#pragma unroll
  for (int j = 0; j < 4; ++j) {
    const int n = nblk + j * 16 + nn;
    const float c = cs[n] * scale;
#pragma unroll
    for (int r = 0; r < 8; ++r) {
      const int m = m0 + mo + r;
      float v = acc[j][r] * c * D[(size_t)m * SEQ + n];
      v = fmaxf(fabsf(v), 1.0f);
      Ah[(size_t)m * SEQ + n] = (_Float16)v;
    }
  }
}

// ---------------------------------------------------------------------------
// out[b,q,d] = sum_k A[q,k] * V[b,k,d]  (B operand = Vt[b][d][k], col-major)
// Same LDS double-buffered pipeline.
// ---------------------------------------------------------------------------
__global__ __launch_bounds__(256) void out_kernel(
    const _Float16* __restrict__ Ah, const _Float16* __restrict__ Vt,
    float* __restrict__ out) {
  __shared__ _Float16 As[2][128 * A_ST];
  __shared__ _Float16 Bs[2][64 * B_ST];

  const int tid  = threadIdx.x;
  const int lane = tid & 31;
  const int wave = tid >> 5;
  const int mblk = blockIdx.x * 128;
  const int nblk = blockIdx.y * 64;
  const int b    = blockIdx.z;
  const _Float16* Vb = Vt + (size_t)b * DIM * SEQ;

  v8f acc[4];
#pragma unroll
  for (int j = 0; j < 4; ++j) acc[j] = vzero8();

  const int NSTEP = SEQ / KSTEP;  // 64

  stage_copy(As[0], Bs[0],
             Ah + (size_t)mblk * SEQ, SEQ,
             Vb + (size_t)nblk * SEQ, SEQ, tid);
  stage_join();

  for (int s = 0; s < NSTEP; ++s) {
    const int cur = s & 1;
    if (s + 1 < NSTEP) {
      stage_copy(As[cur ^ 1], Bs[cur ^ 1],
                 Ah + (size_t)mblk * SEQ + (s + 1) * KSTEP, SEQ,
                 Vb + (size_t)nblk * SEQ + (s + 1) * KSTEP, SEQ, tid);
    }
    v16h a = lds_a_frag(As[cur], wave, lane);
#pragma unroll
    for (int j = 0; j < 4; ++j) {
      v16h bf = lds_b_frag(Bs[cur], j, lane);
      acc[j] = wmma_f16(a, bf, acc[j]);
    }
    stage_join();
  }

  const int m0 = mblk + wave * 16;
  const int mo = (lane >> 4) * 8;
  const int nn = lane & 15;
  float* O = out + (size_t)b * SEQ * DIM;
#pragma unroll
  for (int j = 0; j < 4; ++j)
#pragma unroll
    for (int r = 0; r < 8; ++r)
      O[(size_t)(m0 + mo + r) * DIM + (nblk + j * 16 + nn)] = acc[j][r];
}

// ---------------------------------------------------------------------------
extern "C" void kernel_launch(void* const* d_in, const int* in_sizes, int n_in,
                              void* d_out, int out_size, void* d_ws,
                              size_t ws_size, hipStream_t stream) {
  (void)in_sizes; (void)n_in; (void)out_size; (void)ws_size;
  const float* xq = (const float*)d_in[0];
  const float* xk = (const float*)d_in[1];
  const float* xv = (const float*)d_in[2];
  const float* W  = (const float*)d_in[3];
  const float* D  = (const float*)d_in[4];
  float* out = (float*)d_out;

  const size_t QS = (size_t)BATCH * SEQ * DIM;
  _Float16* Qh = (_Float16*)d_ws;               // [B*S, DIM] row-major
  _Float16* Kh = Qh + QS;                       // [B*S, DIM] row-major
  _Float16* Vt = Kh + QS;                       // [B][DIM][SEQ]
  _Float16* Wt = Vt + QS;                       // [DIM][DIN]
  _Float16* Ah = Wt + (size_t)DIN * DIM;        // [SEQ, SEQ]
  float*    cs = (float*)(Ah + (size_t)SEQ * SEQ);

  transpose_w_kernel<<<(DIN * DIM) / 256, 256, 0, stream>>>(W, Wt);
  colscale_kernel<<<SEQ, 256, 0, stream>>>(D, cs);

  dim3 gp((BATCH * SEQ) / 128, DIM / 64, 3);
  proj_kernel<<<gp, 256, 0, stream>>>(xq, xk, xv, Wt, Qh, Kh, Vt);

  dim3 gs(SEQ / 128, SEQ / 64);
  score_kernel<<<gs, 256, 0, stream>>>(Qh, Kh, D, cs, Ah);

  dim3 go(SEQ / 128, DIM / 64, BATCH);
  out_kernel<<<go, 256, 0, stream>>>(Ah, Vt, out);
}
